// MLFMBertClassifier_64493228917340
// MI455X (gfx1250) — matrix operand order repaired and checked by reading, not compiled, not run on hardware
//
#include <hip/hip_runtime.h>
#include <hip/hip_bf16.h>
#include <stdint.h>

// Shapes (fixed by the harness)
#define Bsz 16
#define Sq  256
#define Hd  768
#define Dd  64
#define Mrows (Bsz * Sq)      // 4096

// GEMM tiling
#define BM 128
#define BN 64
#define BK 64
#define APAD 72               // padded LDS row (halfs): conflict-free frag reads

typedef __bf16 bf16x16 __attribute__((ext_vector_type(16)));
typedef float  f32x8   __attribute__((ext_vector_type(8)));

__device__ __forceinline__ __bf16 f2bf(float f) {
  union { float f; uint32_t u; } x; x.f = f;
  uint32_t r = x.u + 0x7FFFu + ((x.u >> 16) & 1u);   // RNE
  union { uint16_t s; __bf16 b; } y; y.s = (uint16_t)(r >> 16);
  return y.b;
}

__device__ __forceinline__ uint32_t pk2bf(float x, float y) {   // [y|x] packed bf16
  union { float f; uint32_t u; } a, b; a.f = x; b.f = y;
  uint32_t ra = a.u + 0x7FFFu + ((a.u >> 16) & 1u);
  uint32_t rb = b.u + 0x7FFFu + ((b.u >> 16) & 1u);
  return (ra >> 16) | ((rb >> 16) << 16);
}

// Per-thread register staging for one (BMxBK A, BKxBN B) tile
struct TileRegs { float4 a[8]; float4 b[4]; };

__device__ __forceinline__ void load_tile(TileRegs& tr,
                                          const float* __restrict__ A, int K, int rol,
                                          const float* __restrict__ W,
                                          int bm, int bn, int k0, int tid)
{
  #pragma unroll
  for (int it = 0; it < 8; ++it) {            // BM*BK/4 = 2048 float4
    int idx = tid + it * 256;
    int r   = idx >> 4;                        // 16 float4 per 64-wide row
    int c4  = (idx & 15) << 2;
    int gr  = bm + r;
    if (rol) gr = (gr & ~(Sq - 1)) | (((gr & (Sq - 1)) + 1) & (Sq - 1));
    tr.a[it] = *(const float4*)(A + (size_t)gr * K + k0 + c4);
  }
  #pragma unroll
  for (int it = 0; it < 4; ++it) {            // BK*BN/4 = 1024 float4
    int idx = tid + it * 256;
    int k   = idx >> 4;                        // 16 float4 per 64-wide n-row
    int n4  = (idx & 15) << 2;
    tr.b[it] = *(const float4*)(W + (size_t)(k0 + k) * Hd + bn + n4);
  }
}

__device__ __forceinline__ void store_tile(const TileRegs& tr,
                                           __bf16 (*As)[APAD], __bf16 (*Bs)[APAD],
                                           int tid)
{
  #pragma unroll
  for (int it = 0; it < 8; ++it) {
    int idx = tid + it * 256;
    int r   = idx >> 4;
    int c4  = (idx & 15) << 2;
    uint32_t* p = (uint32_t*)&As[r][c4];       // 4-byte aligned (c4 even)
    p[0] = pk2bf(tr.a[it].x, tr.a[it].y);
    p[1] = pk2bf(tr.a[it].z, tr.a[it].w);
  }
  #pragma unroll
  for (int it = 0; it < 4; ++it) {             // transposed scatter: [n][k]
    int idx = tid + it * 256;
    int k   = idx >> 4;
    int n4  = (idx & 15) << 2;
    Bs[n4 + 0][k] = f2bf(tr.b[it].x);
    Bs[n4 + 1][k] = f2bf(tr.b[it].y);
    Bs[n4 + 2][k] = f2bf(tr.b[it].z);
    Bs[n4 + 3][k] = f2bf(tr.b[it].w);
  }
}

// 2 K-steps of 32, 4 N-subtiles -> 8 WMMAs per LDS stage
__device__ __forceinline__ void compute_stage(f32x8 acc[4],
                                              const __bf16 (*As)[APAD],
                                              const __bf16 (*Bs)[APAD],
                                              int wave, int lane)
{
  const int hi  = lane >> 4;
  const int l15 = lane & 15;
  #pragma unroll
  for (int ks = 0; ks < 2; ++ks) {
    // A fragment 16x32 (ISA layout: lanes0-15 K0-7/16-23, lanes16-31 +8)
    bf16x16 afrag;
    const __bf16* ap = &As[(wave << 4) + l15][ks * 32 + (hi << 3)];
    #pragma unroll
    for (int e = 0; e < 8; ++e) afrag[e]     = ap[e];
    #pragma unroll
    for (int e = 0; e < 8; ++e) afrag[8 + e] = ap[16 + e];
    // B fragments 32x16 (lanes0-15 K0-7/16-23, lanes16-31 +8)
    #pragma unroll
    for (int sN = 0; sN < 4; ++sN) {
      bf16x16 bfrag;
      const __bf16* bp = &Bs[(sN << 4) + l15][ks * 32 + (hi << 3)];
      #pragma unroll
      for (int e = 0; e < 16; ++e) bfrag[e] = bp[(e < 8) ? e : (e + 8)];
      acc[sN] = __builtin_amdgcn_wmma_f32_16x16x32_bf16(
          false, afrag, false, bfrag, (short)0, acc[sN], false, false);
    }
  }
}

// ---------------------------------------------------------------------------
// Dual-source bf16-WMMA GEMM:  C[M,768] = A1[M,K1]@W1 + A2[M,K2]@W2 (+bias)
// A2 rows optionally rolled: row (b,i) reads source row (b,(i+1)%S)  [zs]
// Block tile 128x64, 8 waves, wave w -> rows [16w,16w+16) x 64 cols.
// Software-pipelined: next tile's global loads issue before compute stage.
// ---------------------------------------------------------------------------
__global__ __launch_bounds__(256)
void gemm_dual_kernel(const float* __restrict__ A1, int K1,
                      const float* __restrict__ W1,
                      const float* __restrict__ A2, int K2,
                      const float* __restrict__ W2,
                      int rollA2,
                      const float* __restrict__ bias,
                      float* __restrict__ C)
{
  __shared__ __bf16 As[BM][APAD];
  __shared__ __bf16 Bs[BN][APAD];

  const int tid  = threadIdx.x;
  const int lane = tid & 31;
  const int wave = tid >> 5;
  const int bm   = blockIdx.x * BM;
  const int bn   = blockIdx.y * BN;
  const int hi   = lane >> 4;
  const int l15  = lane & 15;

  f32x8 acc[4];
  #pragma unroll
  for (int s = 0; s < 4; ++s)
    #pragma unroll
    for (int e = 0; e < 8; ++e) acc[s][e] = 0.0f;

  #pragma unroll 1
  for (int src = 0; src < 2; ++src) {
    const float* A = src ? A2 : A1;
    const float* W = src ? W2 : W1;
    const int    K = src ? K2 : K1;
    const int  rol = src ? rollA2 : 0;
    if (A == nullptr || K == 0) continue;

    TileRegs tr;
    load_tile(tr, A, K, rol, W, bm, bn, 0, tid);
    store_tile(tr, As, Bs, tid);
    __syncthreads();

    #pragma unroll 1
    for (int k0 = BK; k0 < K; k0 += BK) {
      load_tile(tr, A, K, rol, W, bm, bn, k0, tid);   // overlap with WMMAs
      compute_stage(acc, As, Bs, wave, lane);
      __syncthreads();
      store_tile(tr, As, Bs, tid);
      __syncthreads();
    }
    compute_stage(acc, As, Bs, wave, lane);
    __syncthreads();                                   // LDS reuse by next source
  }

  // ---- epilogue: C layout (VGPR r: lanes0-15 M=r, lanes16-31 M=r+8) ----
  float bv[4] = {0.f, 0.f, 0.f, 0.f};
  if (bias) {
    #pragma unroll
    for (int sN = 0; sN < 4; ++sN) bv[sN] = bias[bn + (sN << 4) + l15];
  }
  #pragma unroll
  for (int sN = 0; sN < 4; ++sN) {
    #pragma unroll
    for (int r = 0; r < 8; ++r) {
      int gm = bm + (wave << 4) + r + (hi << 3);
      int gn = bn + (sN << 4) + l15;
      C[(size_t)gm * Hd + gn] = acc[sN][r] + bv[sN];
    }
  }
}

// ---------------------------------------------------------------------------
// a1[m] = dot(zs[m], w_att[0:768]),  a2[m] = dot(zs[m], w_att[768:1536])
// zs[m=(b,i)] = z[b, (i+1)%S]
// ---------------------------------------------------------------------------
__global__ __launch_bounds__(256)
void k_a12(const float* __restrict__ z, const float* __restrict__ w_att,
           float* __restrict__ a1, float* __restrict__ a2)
{
  __shared__ float r1[256], r2[256];
  const int m = blockIdx.x, t = threadIdx.x;
  const int b = m >> 8, i = m & 255;
  const float* zr = z + (size_t)((b << 8) | ((i + 1) & 255)) * Hd;
  float p1 = 0.f, p2 = 0.f;
  #pragma unroll
  for (int q = 0; q < 3; ++q) {
    int h = t + q * 256;
    float zv = zr[h];
    p1 += zv * w_att[h];
    p2 += zv * w_att[Hd + h];
  }
  r1[t] = p1; r2[t] = p2; __syncthreads();
  for (int o = 128; o > 0; o >>= 1) {
    if (t < o) { r1[t] += r1[t + o]; r2[t] += r2[t + o]; }
    __syncthreads();
  }
  if (t == 0) { a1[m] = r1[0]; a2[m] = r2[0]; }
}

// ---------------------------------------------------------------------------
// Fused attention: scores + leaky-relu + mask + softmax + agg_z + agg_d,
// plus has_nbr block-OR. One block per (b,i) row; thread t == column j.
// ---------------------------------------------------------------------------
__global__ __launch_bounds__(256)
void k_attn(const float* __restrict__ z, const float* __restrict__ dep,
            const int* __restrict__ deprel, const float* __restrict__ w3,
            const float* __restrict__ b_att,
            const float* __restrict__ a1, const float* __restrict__ a2,
            float* __restrict__ aggz, float* __restrict__ aggd,
            int* __restrict__ has_nbr)
{
  __shared__ float red[256];
  __shared__ float attn[256];
  __shared__ int   anyn;

  const int m = blockIdx.x, t = threadIdx.x;
  const int b = m >> 8;

  // score for column j = t
  const float4* dr  = (const float4*)(dep + ((size_t)m * Sq + t) * Dd);
  const float4* w3v = (const float4*)w3;
  float s3 = 0.f;
  #pragma unroll
  for (int q = 0; q < 16; ++q) {
    float4 a = dr[q], w = w3v[q];
    s3 += a.x * w.x + a.y * w.y + a.z * w.z + a.w * w.w;
  }
  const bool msk = deprel[(size_t)m * Sq + t] > 0;
  float sv = a1[m] + a2[(b << 8) | t] + s3 + b_att[0];
  sv = (sv >= 0.f) ? sv : 0.01f * sv;              // leaky relu
  sv = msk ? sv : -1.0e9f;                          // adjacency mask

  if (t == 0) anyn = 0;
  __syncthreads();
  if (msk) anyn = 1;                                // benign race (writes 1)

  // softmax over j
  red[t] = sv; __syncthreads();
  for (int o = 128; o > 0; o >>= 1) {
    if (t < o) red[t] = fmaxf(red[t], red[t + o]);
    __syncthreads();
  }
  const float mx = red[0]; __syncthreads();
  const float ev = __expf(sv - mx);
  red[t] = ev; __syncthreads();
  for (int o = 128; o > 0; o >>= 1) {
    if (t < o) red[t] += red[t + o];
    __syncthreads();
  }
  attn[t] = ev * (1.f / red[0]);
  if (t == 0) has_nbr[m] = anyn;
  __syncthreads();

  // agg_z[m, h] = sum_j attn[j] * zs[b, j, h]   (threads own h -> coalesced)
  float acc0 = 0.f, acc1 = 0.f, acc2 = 0.f;
  const float* zb = z + (size_t)(b << 8) * Hd;
  #pragma unroll 4
  for (int j = 0; j < Sq; ++j) {
    const float w = attn[j];
    const float* zr = zb + (size_t)((j + 1) & 255) * Hd;
    acc0 += w * zr[t];
    acc1 += w * zr[t + 256];
    acc2 += w * zr[t + 512];
  }
  float* az = aggz + (size_t)m * Hd;
  az[t] = acc0; az[t + 256] = acc1; az[t + 512] = acc2;

  // agg_d[m, d] = sum_j attn[j] * dep[m, j, d]  (threads own d -> coalesced)
  const int d = t & 63, jc = t >> 6;
  const float* dpb = dep + (size_t)m * Sq * Dd;
  float accd = 0.f;
  #pragma unroll 4
  for (int jj = 0; jj < 64; ++jj) {
    const int j = (jc << 6) + jj;
    accd += attn[j] * dpb[(size_t)j * Dd + d];
  }
  __syncthreads();
  red[t] = accd; __syncthreads();
  if (t < 128) red[t] += red[t + 128];
  __syncthreads();
  if (t < 64) aggd[(size_t)m * Dd + t] = red[t] + red[t + 64];
}

// ---------------------------------------------------------------------------
// out[b,i,:] = roll(write,1)[b,i] ? roll(temp,1)[b,i,:] : X[b,i,:]
// write[b,i] = (asp_start[b] <= i <= asp_end[b]) && has_nbr[b,i]
// ---------------------------------------------------------------------------
__global__ __launch_bounds__(256)
void k_out(const float* __restrict__ X, const float* __restrict__ temp,
           const int* __restrict__ has_nbr,
           const int* __restrict__ asp_s, const int* __restrict__ asp_e,
           float* __restrict__ out)
{
  const int m = blockIdx.x, t = threadIdx.x;
  const int b = m >> 8, i = m & 255;
  const int ip = (i + 255) & 255;
  const bool cond = (ip >= asp_s[b]) && (ip <= asp_e[b]) &&
                    (has_nbr[(b << 8) | ip] != 0);
  const float* srcv = cond ? (temp + (size_t)((b << 8) | ip) * Hd)
                           : (X + (size_t)m * Hd);
  float* o = out + (size_t)m * Hd;
  o[t]       = srcv[t];
  o[t + 256] = srcv[t + 256];
  o[t + 512] = srcv[t + 512];
}

// ---------------------------------------------------------------------------
extern "C" void kernel_launch(void* const* d_in, const int* in_sizes, int n_in,
                              void* d_out, int out_size, void* d_ws, size_t ws_size,
                              hipStream_t stream)
{
  const float* X      = (const float*)d_in[0];   // (B,S,H)
  const float* dep    = (const float*)d_in[1];   // (B,S,S,D)
  const int*   deprel = (const int*)  d_in[2];   // (B,S,S)
  const int*   asp_s  = (const int*)  d_in[3];
  const int*   asp_e  = (const int*)  d_in[4];
  const float* Wz     = (const float*)d_in[5];   // (H,H)
  const float* bz     = (const float*)d_in[6];   // (H,)
  const float* w_att  = (const float*)d_in[7];   // (2H+D,)
  const float* b_att  = (const float*)d_in[8];   // scalar
  const float* Wf     = (const float*)d_in[9];   // (H+D,H)
  const float* Wh     = (const float*)d_in[10];  // (2H,H)
  float* out = (float*)d_out;

  // workspace carve-out (floats)
  const size_t MH = (size_t)Mrows * Hd;          // 3,145,728
  float* ws   = (float*)d_ws;
  float* z    = ws;                               // MH
  float* aggz = z    + MH;                        // MH
  float* nrep = aggz + MH;                        // MH
  float* temp = nrep + MH;                        // MH
  float* aggd = temp + MH;                        // Mrows*Dd
  float* a1   = aggd + (size_t)Mrows * Dd;        // Mrows
  float* a2   = a1   + Mrows;                     // Mrows
  int*   hn   = (int*)(a2 + Mrows);               // Mrows

  const dim3 gg(Mrows / BM, Hd / BN);             // (32, 12)
  const dim3 gb(256);

  // 1) z = X @ Wz + bz
  gemm_dual_kernel<<<gg, gb, 0, stream>>>(X, Hd, Wz,
                                          nullptr, 0, nullptr, 0,
                                          bz, z);
  // 2) a1 = zs@w1, a2 = zs@w2
  k_a12<<<Mrows, gb, 0, stream>>>(z, w_att, a1, a2);
  // 3) fused attention -> agg_z, agg_d, has_nbr
  k_attn<<<Mrows, gb, 0, stream>>>(z, dep, deprel, w_att + 2 * Hd, b_att,
                                   a1, a2, aggz, aggd, hn);
  // 4) nrep = agg_z @ Wf[:H] + agg_d @ Wf[H:]
  gemm_dual_kernel<<<gg, gb, 0, stream>>>(aggz, Hd, Wf,
                                          aggd, Dd, Wf + (size_t)Hd * Hd, 0,
                                          nullptr, nrep);
  // 5) temp = nrep @ Wh[:H] + zs @ Wh[H:]   (A2 = z with roll)
  gemm_dual_kernel<<<gg, gb, 0, stream>>>(nrep, Hd, Wh,
                                          z, Hd, Wh + (size_t)Hd * Hd, 1,
                                          nullptr, temp);
  // 6) masked roll-write
  k_out<<<Mrows, gb, 0, stream>>>(X, temp, hn, asp_s, asp_e, out);
}
